// FP8Linear_3470333575918
// MI455X (gfx1250) — 8-bit matrix operand order repaired and checked by reading, compiled, not run on hardware
//
#include <hip/hip_runtime.h>
#include <hip/hip_bf16.h>
#include <stdint.h>

typedef __attribute__((ext_vector_type(16))) int   v16i;
typedef __attribute__((ext_vector_type(8)))  float v8f;
typedef __attribute__((ext_vector_type(4)))  unsigned int u32x4;
typedef __attribute__((ext_vector_type(4)))  int   i32x4;
typedef __attribute__((ext_vector_type(4)))  float f32x4;

#define FP8_MAX 448.0f
#define KSTEP   128
#define BLK_M   256
#define BLK_N   64
#define B_TILE_BYTES 8192


__device__ inline unsigned int e4m3_sw(float f) {
    unsigned int u = __float_as_uint(f);
    unsigned int sign = (u >> 31) & 1u;
    int exp = (int)((u >> 23) & 0xFF) - 127;
    unsigned int man = u & 0x7FFFFFu;
    unsigned int out;
    if (((u >> 23) & 0xFF) == 0xFF) { out = 0x7F; }
    else if (exp < -10) { out = 0; }
    else if (exp < -6) {
        unsigned int m = man | 0x800000u;
        int rshift = 21 + (-6 - exp);
        unsigned int r = m >> rshift;
        unsigned int rem = m & ((1u << rshift) - 1u);
        unsigned int half = 1u << (rshift - 1);
        if (rem > half || (rem == half && (r & 1u))) r++;
        out = r;
    } else {
        unsigned int r = man >> 20;
        unsigned int rem = man & 0xFFFFFu;
        if (rem > 0x80000u || (rem == 0x80000u && (r & 1u))) r++;
        int e8 = exp + 7;
        if (r == 8) { r = 0; e8++; }
        if (e8 > 15 || (e8 == 15 && r > 6)) { e8 = 15; r = 6; }
        out = ((unsigned int)e8 << 3) | r;
    }
    return out | (sign << 7);
}

__device__ inline unsigned int pack4_fp8(float a, float b, float c, float d) {
#if __has_builtin(__builtin_amdgcn_cvt_pk_fp8_f32)
    int v = 0;
    v = __builtin_amdgcn_cvt_pk_fp8_f32(a, b, v, false);
    v = __builtin_amdgcn_cvt_pk_fp8_f32(c, d, v, true);
    return (unsigned int)v;
#else
    return e4m3_sw(a) | (e4m3_sw(b) << 8) | (e4m3_sw(c) << 16) | (e4m3_sw(d) << 24);
#endif
}

__device__ __forceinline__ unsigned int quant_dword(f32x4 v, float s) {
    float a = fminf(fmaxf(v.x * s, -FP8_MAX), FP8_MAX);
    float b = fminf(fmaxf(v.y * s, -FP8_MAX), FP8_MAX);
    float c = fminf(fmaxf(v.z * s, -FP8_MAX), FP8_MAX);
    float d = fminf(fmaxf(v.w * s, -FP8_MAX), FP8_MAX);
    return pack4_fp8(a, b, c, d);
}


__global__ __launch_bounds__(256)
void fp8_quant_kernel(const float* __restrict__ in, unsigned int* __restrict__ outq,
                      long n4) {
    long i = (long)blockIdx.x * blockDim.x + threadIdx.x;
    if (i >= n4) return;
    outq[i] = quant_dword(((const f32x4*)in)[i], 1.0f);
}


__global__ __launch_bounds__(256)
void fp8_quant_swizzleA_kernel(const float* __restrict__ in,
                               unsigned int* __restrict__ outq,
                               const float* __restrict__ scale_ptr, int K) {
    const int m  = blockIdx.y;
    const int kd = blockIdx.x * blockDim.x + threadIdx.x;
    const int K4 = K >> 2;
    if (kd >= K4) return;
    const float s = 1.0f / scale_ptr[0];
    f32x4 v = ((const f32x4*)(in + (size_t)m * K))[kd];
    unsigned int packed = quant_dword(v, s);

    const int ks  = kd >> 5;
    const int kr  = kd & 31;
    const int h64 = kr >> 4;
    const int rem = kr & 15;
    const int p   = rem >> 2;
    const int hi  = (kr >> 2) & 1;
    const int q   = rem & 1;
    const int j   = 4 * (kr >> 3) + (kr & 3);
    const int lane = (m & 15) + 16 * hi;
    const size_t frag = ((size_t)(m >> 4) * (K >> 7) + ks) * 32 + lane;
    outq[frag * 16 + j] = packed;
}


#if __has_builtin(__builtin_amdgcn_global_load_async_to_lds_b128)
#define HAVE_ASYNC_LDS 1
#else
#define HAVE_ASYNC_LDS 0
#endif

__device__ __forceinline__ void copy_b128_to_lds(const uint8_t* g, uint8_t* l) {
#if HAVE_ASYNC_LDS
    __builtin_amdgcn_global_load_async_to_lds_b128(
        (__attribute__((address_space(1))) i32x4*)(uint8_t*)g,
        (__attribute__((address_space(3))) i32x4*)l, 0, 0);
#else
    *(u32x4*)l = *(const u32x4*)g;
#endif
}

__device__ __forceinline__ void wait_async_copies() {
#if HAVE_ASYNC_LDS
# if __has_builtin(__builtin_amdgcn_s_wait_asynccnt)
    __builtin_amdgcn_s_wait_asynccnt(0);
# else
    asm volatile("s_wait_asynccnt 0x0" ::: "memory");
# endif
#endif
}


__global__ __launch_bounds__(256)
void fp8_gemm_wmma_kernel(const uint8_t* __restrict__ xqs, const uint8_t* __restrict__ wq,
                          const float* __restrict__ bias,
                          const float* __restrict__ input_scale,
                          const float* __restrict__ weight_scale,
                          __hip_bfloat16* __restrict__ out,
                          int M, int N, int K) {
    __shared__ __align__(16) uint8_t ldsB[2][B_TILE_BYTES];

    const int tid  = threadIdx.x;
    const int wave = tid >> 5;
    const int lane = tid & 31;
    const int hi   = lane >> 4;
    const int ln   = lane & 15;

    const int mWave = blockIdx.y * BLK_M + wave * 32;
    const int nBase = blockIdx.x * BLK_N;
    const int S     = K / KSTEP;

    const int f0 = tid, f1 = tid + 256;
    const int l0 = f0 & 31, g0 = (f0 >> 5) & 3, j0 = f0 >> 7;
    const int l1 = f1 & 31, g1 = (f1 >> 5) & 3, j1 = f1 >> 7;
    const uint8_t* gB0 = wq + (size_t)(nBase + j0 * 16 + (l0 & 15)) * K
                            + 32 * g0 + 16 * (l0 >> 4);
    const uint8_t* gB1 = wq + (size_t)(nBase + j1 * 16 + (l1 & 15)) * K
                            + 32 * g1 + 16 * (l1 >> 4);

    const size_t aTileStride = (size_t)S * 2048;
    const uint8_t* aPtr0 = xqs + (size_t)(mWave >> 4) * aTileStride + lane * 64;
    const uint8_t* aPtr1 = aPtr0 + aTileStride;

    v8f acc[2][4];
    #pragma unroll
    for (int t = 0; t < 2; ++t)
        #pragma unroll
        for (int j = 0; j < 4; ++j)
            acc[t][j] = (v8f){0.f, 0.f, 0.f, 0.f, 0.f, 0.f, 0.f, 0.f};

    auto loadB = [&](const uint8_t* lb, int j, v16i& b) {
        #pragma unroll
        for (int g = 0; g < 4; ++g) {
            u32x4 d = *(const u32x4*)(lb + j * 2048 + g * 512 + lane * 16);
            b[4 * g]     = (int)d.x;
            b[4 * g + 1] = (int)d.y;
            b[4 * g + 2] = (int)d.z;
            b[4 * g + 3] = (int)d.w;
        }
    };

    auto compute_step = [&](int s) {
        v16i a0, a1;
        const uint8_t* ap0 = aPtr0 + (size_t)s * 2048;
        const uint8_t* ap1 = aPtr1 + (size_t)s * 2048;
        #pragma unroll
        for (int g = 0; g < 4; ++g) {
            u32x4 d0 = *(const u32x4*)(ap0 + g * 16);
            u32x4 d1 = *(const u32x4*)(ap1 + g * 16);
            a0[4 * g] = (int)d0.x; a0[4 * g + 1] = (int)d0.y;
            a0[4 * g + 2] = (int)d0.z; a0[4 * g + 3] = (int)d0.w;
            a1[4 * g] = (int)d1.x; a1[4 * g + 1] = (int)d1.y;
            a1[4 * g + 2] = (int)d1.z; a1[4 * g + 3] = (int)d1.w;
        }
        const uint8_t* lb = &ldsB[s & 1][0];
        v16i b0, b1;
        loadB(lb, 0, b0);
        loadB(lb, 1, b1);
        acc[0][0] = __builtin_amdgcn_wmma_f32_16x16x128_fp8_fp8(a0, b0, (short)0, acc[0][0], false, false);
        acc[1][0] = __builtin_amdgcn_wmma_f32_16x16x128_fp8_fp8(a1, b0, (short)0, acc[1][0], false, false);
        loadB(lb, 2, b0);
        acc[0][1] = __builtin_amdgcn_wmma_f32_16x16x128_fp8_fp8(a0, b1, (short)0, acc[0][1], false, false);
        acc[1][1] = __builtin_amdgcn_wmma_f32_16x16x128_fp8_fp8(a1, b1, (short)0, acc[1][1], false, false);
        loadB(lb, 3, b1);
        acc[0][2] = __builtin_amdgcn_wmma_f32_16x16x128_fp8_fp8(a0, b0, (short)0, acc[0][2], false, false);
        acc[1][2] = __builtin_amdgcn_wmma_f32_16x16x128_fp8_fp8(a1, b0, (short)0, acc[1][2], false, false);
        acc[0][3] = __builtin_amdgcn_wmma_f32_16x16x128_fp8_fp8(a0, b1, (short)0, acc[0][3], false, false);
        acc[1][3] = __builtin_amdgcn_wmma_f32_16x16x128_fp8_fp8(a1, b1, (short)0, acc[1][3], false, false);
    };

    copy_b128_to_lds(gB0, &ldsB[0][f0 * 16]);
    copy_b128_to_lds(gB1, &ldsB[0][f1 * 16]);

    for (int s = 0; s < S - 1; ++s) {
        wait_async_copies();
        __syncthreads();
        const int k1 = (s + 1) * KSTEP;
        copy_b128_to_lds(gB0 + k1, &ldsB[(s + 1) & 1][f0 * 16]);
        copy_b128_to_lds(gB1 + k1, &ldsB[(s + 1) & 1][f1 * 16]);
        compute_step(s);
    }
    wait_async_copies();
    __syncthreads();
    compute_step(S - 1);

    const float sc = input_scale[0] * weight_scale[0];
    #pragma unroll
    for (int t = 0; t < 2; ++t) {
        #pragma unroll
        for (int j = 0; j < 4; ++j) {
            const int n = nBase + j * 16 + ln;
            const float bv = bias[n];
            #pragma unroll
            for (int r = 0; r < 8; ++r) {
                const int m = mWave + t * 16 + r + 8 * hi;
                float v = acc[t][j][r] * sc + bv;
                out[(size_t)m * N + n] = __float2bfloat16(v);
            }
        }
    }
}


extern "C" void kernel_launch(void* const* d_in, const int* in_sizes, int n_in,
                              void* d_out, int out_size, void* d_ws, size_t ws_size,
                              hipStream_t stream) {
    const float* x      = (const float*)d_in[0];
    const float* w      = (const float*)d_in[1];
    const float* wscale = (const float*)d_in[2];
    const float* iscale = (const float*)d_in[3];
    const float* bias   = (const float*)d_in[4];

    const int N = in_sizes[4];
    const int K = in_sizes[1] / N;
    const int M = in_sizes[0] / K;

    uint8_t* xqs = (uint8_t*)d_ws;
    size_t xq_bytes = ((size_t)M * K + 255) & ~(size_t)255;
    uint8_t* wq = xqs + xq_bytes;

    {
        dim3 grid((K / 4 + 255) / 256, M);
        fp8_quant_swizzleA_kernel<<<grid, 256, 0, stream>>>(
            x, (unsigned int*)xqs, iscale, K);
    }

    long nw4 = (long)N * K / 4;
    fp8_quant_kernel<<<(nw4 + 255) / 256, 256, 0, stream>>>(
        w, (unsigned int*)wq, nw4);

    dim3 grid(N / BLK_N, M / BLK_M);
    fp8_gemm_wmma_kernel<<<grid, 256, 0, stream>>>(
        xqs, wq, bias, iscale, wscale, (__hip_bfloat16*)d_out, M, N, K);
}
